// TextGenerationModel_82291573391464
// MI455X (gfx1250) — compile-verified, matrix operand order checked by reading, not run on hardware
//
#include <hip/hip_runtime.h>
#include <hip/hip_bf16.h>

// ---------------- problem constants ----------------
#define B_DIM 32
#define T_DIM 512
#define V_DIM 4096
#define H_DIM 1024
#define FOURH 4096
#define M_DIM (B_DIM * T_DIM)   // 16384 rows for the big GEMMs
#define KK_CNT 32               // K=1024 / 32 per WMMA

typedef __bf16 bf16_t;
typedef __attribute__((ext_vector_type(16))) __bf16 v16bf;
typedef __attribute__((ext_vector_type(8)))  float  v8f;

// ---- WMMA helpers -------------------------------------------------------
__device__ __forceinline__ v8f wmma_bf16(v16bf a, v16bf b, v8f c) {
  // emits v_wmma_f32_16x16x32_bf16
  return __builtin_amdgcn_wmma_f32_16x16x32_bf16(false, a, false, b, (short)0, c, false, false);
}

// K index of fragment element e (0..15) given lane-half (ISA 7.12.2, 16-bit A 16x32)
__device__ __forceinline__ int frag_k(int half, int e) {
  return (e < 8) ? (8 * half + e) : (16 + 8 * half + (e - 8));
}

// Build an A fragment (16x32 bf16) from one LDS row of 32 contiguous K values.
// Pairs are contiguous: lowers to two ds_load_b128.
__device__ __forceinline__ v16bf load_a_frag_lds(const bf16_t* rowbase, int half) {
  union { v16bf v; unsigned int u[8]; } A;
  const unsigned int* r = (const unsigned int*)rowbase;
#pragma unroll
  for (int p = 0; p < 8; ++p) {
    int kpair = (p < 4) ? (4 * half + p) : (8 + 4 * half + (p - 4));
    A.u[p] = r[kpair];
  }
  return A.v;
}

// Load a pre-packed B fragment: 16 bf16 = 32B = two global_load_b128.
__device__ __forceinline__ v16bf load_b_frag(const bf16_t* pk) {
  union { v16bf v; uint4 q[2]; } Bf;
  const uint4* p = (const uint4*)pk;
  Bf.q[0] = p[0];
  Bf.q[1] = p[1];
  return Bf.v;
}

__device__ __forceinline__ float sigmoidf_(float x) {
  return 1.0f / (1.0f + __expf(-x));
}

// ---------------- pre-pass kernels ----------------

// Transpose 4096x4096 fp32 (W_ih_l0 -> Wt so the embedding gather is row-major)
__global__ void __launch_bounds__(256) transpose4096(const float* __restrict__ src,
                                                     float* __restrict__ dst) {
  __shared__ float tile[32][33];
  int x  = blockIdx.x * 32 + threadIdx.x;
  int y0 = blockIdx.y * 32;
  for (int i = threadIdx.y; i < 32; i += 8)
    tile[i][threadIdx.x] = src[(size_t)(y0 + i) * 4096 + x];
  __syncthreads();
  int xo  = blockIdx.y * 32 + threadIdx.x;
  int yo0 = blockIdx.x * 32;
  for (int i = threadIdx.y; i < 32; i += 8)
    dst[(size_t)(yo0 + i) * 4096 + xo] = tile[threadIdx.x][i];
}

// Pack an fp32 weight matrix into bf16 WMMA B-fragment order:
// out[((ntile*32 + kk)*32 + lane)*16 + e], N=4096, K=1024.
// transposed=1: B(k,n) = W[n*ld + k]   (for X @ W.T)
// transposed=0: B(k,n) = W[k*ld + n]   (for X @ W)
__global__ void __launch_bounds__(256) pack_b_kernel(const float* __restrict__ W,
                                                     bf16_t* __restrict__ out,
                                                     int ld, int transposed) {
  size_t idx = (size_t)blockIdx.x * 256 + threadIdx.x;   // 4096*1024 total
  int e     = (int)(idx & 15);
  int lane  = (int)((idx >> 4) & 31);
  int kk    = (int)((idx >> 9) & 31);
  int ntile = (int)(idx >> 14);
  int half  = lane >> 4;
  int n = ntile * 16 + (lane & 15);
  int k = kk * 32 + frag_k(half, e);
  float v = transposed ? W[(size_t)n * ld + k] : W[(size_t)k * ld + n];
  out[idx] = (bf16_t)v;
}

// Embedding gather: xp0[tok, :] = Wt[x[tok], :] + b_ih_l0
__global__ void __launch_bounds__(256) gather_embed(const int* __restrict__ x,
                                                    const float* __restrict__ Wt,
                                                    const float* __restrict__ b,
                                                    float* __restrict__ xp) {
  int tok = blockIdx.x;
  int xv  = x[tok];
  const float4* src = (const float4*)(Wt + (size_t)xv * 4096);
  const float4* bb  = (const float4*)b;
  float4*       dst = (float4*)(xp + (size_t)tok * 4096);
  for (int j = threadIdx.x; j < 1024; j += 256) {
    float4 a = src[j], c = bb[j];
    a.x += c.x; a.y += c.y; a.z += c.z; a.w += c.w;
    dst[j] = a;
  }
}

__global__ void zero_f32(float* __restrict__ p, int n) {
  int i = blockIdx.x * 256 + threadIdx.x;
  if (i < n) p[i] = 0.0f;
}

// ---------------- recurrent LSTM step ----------------
// One launch = one timestep. 16 blocks x 256 threads (8 waves).
// The ENTIRE h_{t-1} (32x1024 bf16 = 64 KB, well under the 320 KB WGP LDS)
// is staged once -> a single barrier per step; the K-loop runs barrier-free.
// Each wave computes the same 16x16 tile for all four gates so the cell
// update happens elementwise on the v8f accumulators.
__global__ void __launch_bounds__(256)
lstm_step(const float* __restrict__ xp,      // [B,T,4H] gate preacts (incl. b_ih)
          const bf16_t* __restrict__ Wpk,    // packed W_hh, fragment order, N=4H
          const float* __restrict__ b_hh,    // [4H]
          float* __restrict__ cstate,        // [B,H] fp32
          bf16_t* __restrict__ hout,         // [B,T,H] bf16 (also next-step h input)
          int t) {
  __shared__ bf16_t shA[B_DIM * H_DIM];      // full h_{t-1}: 64 KB
  const int tid  = threadIdx.x;
  const int lane = tid & 31;
  const int w    = tid >> 5;
  const int half = lane >> 4;
  const int nl   = lane & 15;
  const int wn   = w & 3;
  const int m0   = (w >> 2) * 16;
  const int bn   = blockIdx.x;
  const int nG   = bn * 64 + wn * 16 + nl;   // hidden column this lane owns
  const int nt0  = bn * 4 + wn;              // gate-local n-tile index (0..63)

  // stage h_{t-1} (or zeros at t==0): 8 threads per batch row, 16 uint4 each
  {
    int rrow = tid >> 3;                     // 0..31
    int rq   = tid & 7;
    uint4* s = (uint4*)shA;                  // 32 rows x 128 uint4
    if (t == 0) {
      uint4 z; z.x = 0u; z.y = 0u; z.z = 0u; z.w = 0u;
#pragma unroll
      for (int q = 0; q < 16; ++q) s[rrow * 128 + rq + q * 8] = z;
    } else {
      const uint4* g =
          (const uint4*)(hout + ((size_t)rrow * T_DIM + (t - 1)) * H_DIM);
#pragma unroll
      for (int q = 0; q < 16; ++q) s[rrow * 128 + rq + q * 8] = g[rq + q * 8];
    }
  }
  __syncthreads();

  v8f zero = {};
  v8f acc[4] = {zero, zero, zero, zero};

  for (int kk = 0; kk < KK_CNT; ++kk) {
    // load all four gate fragments into DISTINCT registers first, so the
    // four WMMAs issue back-to-back behind a single load wait
    v16bf bfrag[4];
#pragma unroll
    for (int g = 0; g < 4; ++g) {
      size_t off = ((((size_t)(g * 64 + nt0)) * KK_CNT + kk) * 32 + lane) * 16;
      bfrag[g] = load_b_frag(Wpk + off);
      if (kk + 1 < KK_CNT)
        __builtin_prefetch(Wpk + off + (size_t)32 * 16, 0, 1);  // next K-slice
    }
    v16bf a = load_a_frag_lds(shA + (m0 + nl) * H_DIM + kk * 32, half);
#pragma unroll
    for (int g = 0; g < 4; ++g) acc[g] = wmma_bf16(a, bfrag[g], acc[g]);
  }

  // cell update directly on accumulators (i/f/g/o tiles share the C layout)
  float bi  = b_hh[nG];
  float bff = b_hh[H_DIM + nG];
  float bg  = b_hh[2 * H_DIM + nG];
  float bo  = b_hh[3 * H_DIM + nG];
#pragma unroll
  for (int r = 0; r < 8; ++r) {
    int b = m0 + 8 * half + r;                       // batch row 0..31
    size_t xb = ((size_t)b * T_DIM + t) * FOURH + nG;
    float gi = acc[0][r] + xp[xb]               + bi;
    float gf = acc[1][r] + xp[xb + H_DIM]       + bff;
    float gg = acc[2][r] + xp[xb + 2 * H_DIM]   + bg;
    float go = acc[3][r] + xp[xb + 3 * H_DIM]   + bo;
    float iv = sigmoidf_(gi);
    float fv = sigmoidf_(gf);
    float gv = tanhf(gg);
    float ov = sigmoidf_(go);
    size_t ci = (size_t)b * H_DIM + nG;
    float c = fv * cstate[ci] + iv * gv;
    cstate[ci] = c;
    float h = ov * tanhf(c);
    hout[((size_t)b * T_DIM + t) * H_DIM + nG] = (bf16_t)h;
  }
}

// ---------------- big GEMM: C[M,4096] = A[M,1024](bf16) x Bpk + bias ------
// Block tile 64x128, 8 waves (2 in M x 4 in N), each wave 2x2 WMMA tiles.
__global__ void __launch_bounds__(256)
gemm_bf16(const bf16_t* __restrict__ A, const bf16_t* __restrict__ Bpk,
          const float* __restrict__ bias, float* __restrict__ C) {
  __shared__ bf16_t shA[64 * 32];            // A K-slice: 64 rows x 32 K (4 KB)
  const int tid  = threadIdx.x;
  const int lane = tid & 31;
  const int w    = tid >> 5;
  const int half = lane >> 4;
  const int nl   = lane & 15;
  const int wm   = w >> 2;                   // 0..1
  const int wn   = w & 3;                    // 0..3
  const int mBase = blockIdx.x * 64;
  const int ntile0 = blockIdx.y * 8 + wn * 2;

  v8f zero = {};
  v8f acc[2][2] = {{zero, zero}, {zero, zero}};

  for (int kk = 0; kk < KK_CNT; ++kk) {
    __syncthreads();
    {   // stage A[mBase..mBase+63, kk*32..+32): one b128 per thread
      int row = tid >> 2, q = tid & 3;
      const uint4* g = (const uint4*)(A + ((size_t)(mBase + row) * H_DIM + kk * 32));
      ((uint4*)shA)[tid] = g[q];
    }
    __syncthreads();

    // both B fragments first (distinct registers), then both A frags, then
    // 4 WMMAs back-to-back
    size_t off0 = (((size_t)ntile0 * KK_CNT + kk) * 32 + lane) * 16;
    size_t off1 = (((size_t)(ntile0 + 1) * KK_CNT + kk) * 32 + lane) * 16;
    v16bf b0 = load_b_frag(Bpk + off0);
    v16bf b1 = load_b_frag(Bpk + off1);
    if (kk + 1 < KK_CNT) {
      __builtin_prefetch(Bpk + off0 + (size_t)32 * 16, 0, 1);
      __builtin_prefetch(Bpk + off1 + (size_t)32 * 16, 0, 1);
    }
    v16bf a0 = load_a_frag_lds(shA + (wm * 32 + nl) * 32, half);
    v16bf a1 = load_a_frag_lds(shA + (wm * 32 + 16 + nl) * 32, half);
    acc[0][0] = wmma_bf16(a0, b0, acc[0][0]);
    acc[1][0] = wmma_bf16(a1, b0, acc[1][0]);
    acc[0][1] = wmma_bf16(a0, b1, acc[0][1]);
    acc[1][1] = wmma_bf16(a1, b1, acc[1][1]);
  }

#pragma unroll
  for (int mt = 0; mt < 2; ++mt) {
#pragma unroll
    for (int ntl = 0; ntl < 2; ++ntl) {
      int n = (ntile0 + ntl) * 16 + nl;
      float bv = bias[n];
#pragma unroll
      for (int r = 0; r < 8; ++r) {
        int m = mBase + wm * 32 + mt * 16 + 8 * half + r;
        C[(size_t)m * FOURH + n] = acc[mt][ntl][r] + bv;
      }
    }
  }
}

// ---------------- host orchestration ----------------
extern "C" void kernel_launch(void* const* d_in, const int* in_sizes, int n_in,
                              void* d_out, int out_size, void* d_ws, size_t ws_size,
                              hipStream_t stream) {
  const int*   x       = (const int*)d_in[0];
  const float* W_ih_l0 = (const float*)d_in[1];
  const float* W_hh_l0 = (const float*)d_in[2];
  const float* b_ih_l0 = (const float*)d_in[3];
  const float* b_hh_l0 = (const float*)d_in[4];
  const float* W_ih_l1 = (const float*)d_in[5];
  const float* W_hh_l1 = (const float*)d_in[6];
  const float* b_ih_l1 = (const float*)d_in[7];
  const float* b_hh_l1 = (const float*)d_in[8];
  const float* Wph     = (const float*)d_in[9];
  const float* bp      = (const float*)d_in[10];
  float* out = (float*)d_out;

  // workspace partition (all offsets 256B aligned)
  char* ws = (char*)d_ws;
  float*  xp    = (float*)ws;                      ws += (size_t)M_DIM * FOURH * 4; // 256MB (xp0 then xp1)
  bf16_t* h1    = (bf16_t*)ws;                     ws += (size_t)M_DIM * H_DIM * 2; // 32MB
  bf16_t* h2    = (bf16_t*)ws;                     ws += (size_t)M_DIM * H_DIM * 2; // 32MB
  float*  Wt    = (float*)ws;                      ws += (size_t)V_DIM * FOURH * 4; // 64MB
  bf16_t* pack0 = (bf16_t*)ws;                     ws += (size_t)FOURH * H_DIM * 2; // 8MB
  bf16_t* pack1 = (bf16_t*)ws;                     ws += (size_t)FOURH * H_DIM * 2;
  bf16_t* packm = (bf16_t*)ws;                     ws += (size_t)FOURH * H_DIM * 2;
  bf16_t* packp = (bf16_t*)ws;                     ws += (size_t)FOURH * H_DIM * 2;
  float*  cst   = (float*)ws;                      ws += (size_t)B_DIM * H_DIM * 4; // 128KB

  const int packBlocks = (FOURH * H_DIM) / 256;    // 16384

  // pre-passes (independent; run back-to-back on the stream)
  transpose4096<<<dim3(128, 128), dim3(32, 8), 0, stream>>>(W_ih_l0, Wt);
  pack_b_kernel<<<packBlocks, 256, 0, stream>>>(W_hh_l0, pack0, H_DIM, 1);
  pack_b_kernel<<<packBlocks, 256, 0, stream>>>(W_hh_l1, pack1, H_DIM, 1);
  pack_b_kernel<<<packBlocks, 256, 0, stream>>>(W_ih_l1, packm, H_DIM, 1);
  pack_b_kernel<<<packBlocks, 256, 0, stream>>>(Wph,     packp, FOURH, 0);
  zero_f32<<<(B_DIM * H_DIM + 255) / 256, 256, 0, stream>>>(cst, B_DIM * H_DIM);

  // embedding gather -> xp0
  gather_embed<<<M_DIM, 256, 0, stream>>>(x, Wt, b_ih_l0, xp);

  // layer 0 scan
  for (int t = 0; t < T_DIM; ++t)
    lstm_step<<<16, 256, 0, stream>>>(xp, pack0, b_hh_l0, cst, h1, t);

  // xp1 = h1 @ W_ih_l1.T + b_ih_l1  (overwrites xp0; xp0 no longer read)
  gemm_bf16<<<dim3(M_DIM / 64, FOURH / 128), 256, 0, stream>>>(h1, packm, b_ih_l1, xp);

  // layer 1 scan
  zero_f32<<<(B_DIM * H_DIM + 255) / 256, 256, 0, stream>>>(cst, B_DIM * H_DIM);
  for (int t = 0; t < T_DIM; ++t)
    lstm_step<<<16, 256, 0, stream>>>(xp, pack1, b_hh_l1, cst, h2, t);

  // logits = h2 @ Wph + bp
  gemm_bf16<<<dim3(M_DIM / 64, V_DIM / 128), 256, 0, stream>>>(h2, packp, bp, out);

  (void)in_sizes; (void)n_in; (void)out_size; (void)ws_size;
}
